// SelfAttentionLayer_25220047962240
// MI455X (gfx1250) — matrix operand order repaired and checked by reading, not compiled, not run on hardware
//
#include <hip/hip_runtime.h>
#include <hip/hip_bf16.h>
#include <stdint.h>

typedef __attribute__((ext_vector_type(16))) __bf16 v16bf;
typedef __attribute__((ext_vector_type(8)))  __bf16 v8bf;
typedef __attribute__((ext_vector_type(8)))  float  v8f;
typedef __attribute__((ext_vector_type(4)))  unsigned int v4u;
typedef __attribute__((ext_vector_type(8)))  int v8i;
typedef __attribute__((ext_vector_type(4)))  int v4i;

#define NB 8
#define SEQ 2048
#define DIM 1024

__device__ __forceinline__ __bf16 f2bf(float f) { return (__bf16)f; }

// generic pointer to LDS: low 32 bits == wave-relative LDS byte address
__device__ __forceinline__ uint32_t lds_off(const void* p) {
  return (uint32_t)(uintptr_t)p;
}

// ---------------------------------------------------------------------------
// WMMA bf16 fragment helpers (CDNA5 wave32 layouts, CDNA5 ISA)
// A (16x32): lane l: row=l&15; halves 0-7 = K[(l>=16)*8..], 8-15 = K[16+(l>=16)*8..]
// B (32x16): lane l: col=l&15; halves 0-7 = K[(l>=16)*8..], 8-15 = K[16+(l>=16)*8..]
// C/D f32:   VGPR r, lane l -> (row = r + 8*(l>=16), col = l&15)
// ---------------------------------------------------------------------------
__device__ __forceinline__ v16bf load_afrag(const __bf16* rowbase, int hf) {
  v8bf a0 = *(const v8bf*)(rowbase + hf * 8);
  v8bf a1 = *(const v8bf*)(rowbase + 16 + hf * 8);
  return __builtin_shufflevector(a0, a1, 0,1,2,3,4,5,6,7,8,9,10,11,12,13,14,15);
}
__device__ __forceinline__ v16bf load_bfrag(const __bf16* colbase, int hf) {
  v8bf b0 = *(const v8bf*)(colbase + hf * 8);
  v8bf b1 = *(const v8bf*)(colbase + 16 + hf * 8);
  return __builtin_shufflevector(b0, b1, 0,1,2,3,4,5,6,7,8,9,10,11,12,13,14,15);
}

// 16(M)x128(N) wave tile, K=32 step; B-fragment loads pipelined one tile ahead
__device__ __forceinline__ void mma_tile(const __bf16* atile, const __bf16* btile,
                                         int w, int hf, int mr, v8f acc[8]) {
  const v16bf afrag = load_afrag(&atile[(w * 16 + mr) * 32], hf);
  v16bf bfrag = load_bfrag(&btile[mr * 32], hf);
  #pragma unroll
  for (int nt = 0; nt < 8; ++nt) {
    const v16bf cur = bfrag;
    if (nt < 7) bfrag = load_bfrag(&btile[((nt + 1) * 16 + mr) * 32], hf);
    acc[nt] = __builtin_amdgcn_wmma_f32_16x16x32_bf16(false, afrag, false, cur,
                                                      (short)0, acc[nt], false, false);
  }
}

// ---------------------------------------------------------------------------
// Async global->LDS copy of a 128x32 bf16 tile (gfx1250 ASYNCcnt path).
// Each thread moves 2x16B chunks; per wave one instr = 32 lanes x 16B.
// ---------------------------------------------------------------------------
__device__ __forceinline__ void async_copy_tile_128x32(uint32_t lds_base,
    const __bf16* gbase, int row_stride_elems, int t) {
  #pragma unroll
  for (int c = t; c < 512; c += 256) {
    const int r = c >> 2, c8 = (c & 3) << 3;
    const uint32_t dst = lds_base + (uint32_t)((r * 32 + c8) << 1);
    const uint64_t src = (uint64_t)(uintptr_t)(gbase + (size_t)r * row_stride_elems + c8);
    asm volatile("global_load_async_to_lds_b128 %0, %1, off"
                 :: "v"(dst), "v"(src) : "memory");
  }
}
__device__ __forceinline__ void wait_async0() {
  asm volatile("s_wait_asynccnt 0x0" ::: "memory");
}

// ---------------------------------------------------------------------------
// Tensor Data Mover: 2D bf16 tile load (CDNA5 ISA D# layout)
// clang-23 toolchain: 6-arg builtin (g0 v4u, g1 v8i, g2 v4i, g3 v4i, v8i, cpol)
// ---------------------------------------------------------------------------
__device__ __forceinline__ void tdm_load_2d_bf16(uint32_t lds_dst, const __bf16* gsrc,
    uint32_t tile_w, uint32_t tile_h, uint32_t tensor_w, uint32_t tensor_h,
    uint32_t row_stride_elems) {
  const uint64_t ga = (uint64_t)(uintptr_t)gsrc;
  v4u g0;
  g0[0] = 1u;                                                // count=1 (valid user D#)
  g0[1] = lds_dst;                                           // lds_addr
  g0[2] = (uint32_t)ga;                                      // global_addr[31:0]
  g0[3] = (uint32_t)((ga >> 32) & 0x1FFFFFFu) | (2u << 30);  // global_addr[56:32]|type=2
  v8i g1;
  g1[0] = (int)(1u << 16);                                   // data_size=1 -> 2 bytes
  g1[1] = (int)((tensor_w & 0xFFFFu) << 16);                 // tensor_dim0[15:0]
  g1[2] = (int)(((tensor_w >> 16) & 0xFFFFu) |
                ((tensor_h & 0xFFFFu) << 16));               // dim0 hi | dim1 lo
  g1[3] = (int)(((tensor_h >> 16) & 0xFFFFu) |
                ((tile_w & 0xFFFFu) << 16));                 // dim1 hi | tile_dim0
  g1[4] = (int)(tile_h & 0xFFFFu);                           // tile_dim1 (tile_dim2=0)
  g1[5] = (int)row_stride_elems;                             // tensor_dim0_stride lo
  g1[6] = 0;
  g1[7] = 0;
  const v4i z4 = {0, 0, 0, 0};
  const v8i z8 = {0, 0, 0, 0, 0, 0, 0, 0};
  __builtin_amdgcn_tensor_load_to_lds(g0, g1, z4, z4, z8, 0);
}

// ---------------------------------------------------------------------------
// Kernel 1: out_bf16[M,N] = (X_f32[M,K] @ W_f32[K,N] + bias) * scale
// (f32->bf16 conversion on the fly -> synchronous staging + prefetch)
// ---------------------------------------------------------------------------
__global__ __launch_bounds__(256)
void qkv_gemm_kernel(const float* __restrict__ X, const float* __restrict__ W,
                     const float* __restrict__ bias, __bf16* __restrict__ out,
                     int K, int N, float scale)
{
  __shared__ __bf16 ldsA[128 * 32];   // [m][k]
  __shared__ __bf16 ldsBt[128 * 32];  // [n][k]
  const int t = threadIdx.x;
  const int w = t >> 5, l = t & 31;
  const int hf = l >> 4, mr = l & 15;
  const int m0 = blockIdx.y * 128, n0 = blockIdx.x * 128;

  v8f acc[8];
  #pragma unroll
  for (int i = 0; i < 8; ++i) acc[i] = v8f{0.f,0.f,0.f,0.f,0.f,0.f,0.f,0.f};

  for (int k0 = 0; k0 < K; k0 += 32) {
    __syncthreads();
    { // A tile: f32 -> bf16
      const int r = t >> 3, kk = (t & 7) << 2;
      #pragma unroll
      for (int rr = 0; rr < 128; rr += 32) {
        const float4 v = *(const float4*)&X[(size_t)(m0 + r + rr) * K + k0 + kk];
        __bf16* p = &ldsA[(r + rr) * 32 + kk];
        p[0] = f2bf(v.x); p[1] = f2bf(v.y); p[2] = f2bf(v.z); p[3] = f2bf(v.w);
      }
      if (k0 + 32 < K)
        __builtin_prefetch(&X[(size_t)(m0 + r) * K + k0 + 32 + kk], 0, 0);
    }
    { // B tile transposed: W rows k, cols n -> ldsBt[n][k]
      const int kr = t >> 5, c4 = (t & 31) << 2;
      #pragma unroll
      for (int kk2 = 0; kk2 < 32; kk2 += 8) {
        const float4 v = *(const float4*)&W[(size_t)(k0 + kr + kk2) * N + n0 + c4];
        ldsBt[(c4 + 0) * 32 + kr + kk2] = f2bf(v.x);
        ldsBt[(c4 + 1) * 32 + kr + kk2] = f2bf(v.y);
        ldsBt[(c4 + 2) * 32 + kr + kk2] = f2bf(v.z);
        ldsBt[(c4 + 3) * 32 + kr + kk2] = f2bf(v.w);
      }
      if (k0 + 32 < K)
        __builtin_prefetch(&W[(size_t)(k0 + 32 + kr) * N + n0 + c4], 0, 0);
    }
    __syncthreads();
    mma_tile(ldsA, ldsBt, w, hf, mr, acc);
  }
  #pragma unroll
  for (int nt = 0; nt < 8; ++nt) {
    const int n = n0 + nt * 16 + mr;
    const float bv = bias[n];
    #pragma unroll
    for (int r = 0; r < 8; ++r) {
      const int m = m0 + w * 16 + hf * 8 + r;
      out[(size_t)m * N + n] = f2bf((acc[nt][r] + bv) * scale);
    }
  }
}

// ---------------------------------------------------------------------------
// Kernel 2: S_f32[b,q,k] = Qb[b,q,:] . Kb[b,k,:]
// double-buffered async global->LDS staging (ASYNCcnt)
// ---------------------------------------------------------------------------
__global__ __launch_bounds__(256)
void scores_kernel(const __bf16* __restrict__ Qb, const __bf16* __restrict__ Kb,
                   float* __restrict__ S)
{
  __shared__ __bf16 ldsQ[2][128 * 32];
  __shared__ __bf16 ldsK[2][128 * 32];
  const int t = threadIdx.x;
  const int w = t >> 5, l = t & 31;
  const int hf = l >> 4, mr = l & 15;
  const int b = blockIdx.z;
  const int q0 = blockIdx.y * 128, k0 = blockIdx.x * 128;
  const __bf16* Qt = Qb + (size_t)b * SEQ * DIM + (size_t)q0 * DIM;
  const __bf16* Kt = Kb + (size_t)b * SEQ * DIM + (size_t)k0 * DIM;

  v8f acc[8];
  #pragma unroll
  for (int i = 0; i < 8; ++i) acc[i] = v8f{0.f,0.f,0.f,0.f,0.f,0.f,0.f,0.f};

  async_copy_tile_128x32(lds_off(&ldsQ[0][0]), Qt, DIM, t);
  async_copy_tile_128x32(lds_off(&ldsK[0][0]), Kt, DIM, t);

  constexpr int T = DIM / 32;
  for (int i = 0; i < T; ++i) {
    const int cur = i & 1;
    wait_async0();
    __syncthreads();
    if (i + 1 < T) {
      async_copy_tile_128x32(lds_off(&ldsQ[cur ^ 1][0]), Qt + (i + 1) * 32, DIM, t);
      async_copy_tile_128x32(lds_off(&ldsK[cur ^ 1][0]), Kt + (i + 1) * 32, DIM, t);
    }
    mma_tile(ldsQ[cur], ldsK[cur], w, hf, mr, acc);
  }
  #pragma unroll
  for (int nt = 0; nt < 8; ++nt) {
    #pragma unroll
    for (int r = 0; r < 8; ++r) {
      const int q = q0 + w * 16 + hf * 8 + r;
      const int k = k0 + nt * 16 + mr;
      S[((size_t)b * SEQ + q) * SEQ + k] = acc[nt][r];
    }
  }
}

// ---------------------------------------------------------------------------
// Kernel 3: masked softmax per row, f32 scores -> bf16 probabilities
// ---------------------------------------------------------------------------
__global__ __launch_bounds__(256)
void softmax_kernel(const float* __restrict__ S, const int* __restrict__ mask,
                    __bf16* __restrict__ P)
{
  __shared__ float row[SEQ];
  __shared__ float red[8];
  const int t = threadIdx.x;
  const int rowid = blockIdx.x;          // b*SEQ + q
  const int b = rowid >> 11;
  const float* srow = S + (size_t)rowid * SEQ;
  const int* mrow = mask + (size_t)b * SEQ;

  float lmax = -3.4e38f;
  for (int i = t; i < SEQ; i += 256) {
    const float s = mrow[i] ? srow[i] : -3.4e38f;
    row[i] = s;
    lmax = fmaxf(lmax, s);
  }
  #pragma unroll
  for (int off = 16; off; off >>= 1) lmax = fmaxf(lmax, __shfl_xor(lmax, off));
  if ((t & 31) == 0) red[t >> 5] = lmax;
  __syncthreads();
  const float m = fmaxf(fmaxf(fmaxf(red[0], red[1]), fmaxf(red[2], red[3])),
                        fmaxf(fmaxf(red[4], red[5]), fmaxf(red[6], red[7])));
  float lsum = 0.f;
  for (int i = t; i < SEQ; i += 256) {
    const float e = __expf(row[i] - m);
    row[i] = e;
    lsum += e;
  }
  #pragma unroll
  for (int off = 16; off; off >>= 1) lsum += __shfl_xor(lsum, off);
  __syncthreads();
  if ((t & 31) == 0) red[t >> 5] = lsum;
  __syncthreads();
  const float inv = 1.f / (((red[0] + red[1]) + (red[2] + red[3])) +
                           ((red[4] + red[5]) + (red[6] + red[7])));
  __bf16* prow = P + (size_t)rowid * SEQ;
  for (int i = t; i < SEQ; i += 256) prow[i] = f2bf(row[i] * inv);
}

// ---------------------------------------------------------------------------
// Kernel 4: Out_f32[b,q,d] = sum_k P[b,q,k] * V[b,k,d]
// P tiles staged by the Tensor Data Mover (TENSORcnt), V transposed manually
// ---------------------------------------------------------------------------
__device__ __forceinline__ void v_transpose_fill(__bf16* dst, const __bf16* Vbase,
                                                 int k0, int n0, int t) {
  const int kr = t >> 4, c8 = (t & 15) << 3;
  #pragma unroll
  for (int kk2 = 0; kk2 < 32; kk2 += 16) {
    const v8bf v = *(const v8bf*)&Vbase[(size_t)(k0 + kr + kk2) * DIM + n0 + c8];
    #pragma unroll
    for (int j = 0; j < 8; ++j)
      dst[(c8 + j) * 32 + kr + kk2] = v[j];
  }
}

__global__ __launch_bounds__(256)
void context_kernel(const __bf16* __restrict__ P, const __bf16* __restrict__ Vb,
                    float* __restrict__ Out)
{
  __shared__ __bf16 ldsP[2][128 * 32];   // [q][k]
  __shared__ __bf16 ldsVt[2][128 * 32];  // [d][k]
  const int t = threadIdx.x;
  const int w = t >> 5, l = t & 31;
  const int hf = l >> 4, mr = l & 15;
  const int b = blockIdx.z;
  const int q0 = blockIdx.y * 128, n0 = blockIdx.x * 128;
  const __bf16* Pt = P + (size_t)b * SEQ * SEQ + (size_t)q0 * SEQ;
  const __bf16* Vbase = Vb + (size_t)b * SEQ * DIM;

  v8f acc[8];
  #pragma unroll
  for (int i = 0; i < 8; ++i) acc[i] = v8f{0.f,0.f,0.f,0.f,0.f,0.f,0.f,0.f};

  if (w == 0)
    tdm_load_2d_bf16(lds_off(&ldsP[0][0]), Pt, 32, 128, SEQ, SEQ, SEQ);
  v_transpose_fill(ldsVt[0], Vbase, 0, n0, t);

  constexpr int T = SEQ / 32;
  for (int i = 0; i < T; ++i) {
    const int cur = i & 1;
    __builtin_amdgcn_s_wait_tensorcnt(0);
    __syncthreads();
    if (i + 1 < T) {
      if (w == 0)
        tdm_load_2d_bf16(lds_off(&ldsP[cur ^ 1][0]), Pt + (i + 1) * 32,
                         32, 128, SEQ, SEQ, SEQ);
      v_transpose_fill(ldsVt[cur ^ 1], Vbase, (i + 1) * 32, n0, t);
    }
    mma_tile(ldsP[cur], ldsVt[cur], w, hf, mr, acc);
  }
  #pragma unroll
  for (int nt = 0; nt < 8; ++nt) {
    #pragma unroll
    for (int r = 0; r < 8; ++r) {
      const int q = q0 + w * 16 + hf * 8 + r;
      const int d = n0 + nt * 16 + mr;
      Out[((size_t)b * SEQ + q) * DIM + d] = acc[nt][r];
    }
  }
}

// ---------------------------------------------------------------------------
extern "C" void kernel_launch(void* const* d_in, const int* in_sizes, int n_in,
                              void* d_out, int out_size, void* d_ws, size_t ws_size,
                              hipStream_t stream) {
  const float* x    = (const float*)d_in[0];
  const int*   mask = (const int*)  d_in[1];
  const float* Wq   = (const float*)d_in[2];
  const float* bq   = (const float*)d_in[3];
  const float* Wk   = (const float*)d_in[4];
  const float* bk   = (const float*)d_in[5];
  const float* Wv   = (const float*)d_in[6];
  const float* bv   = (const float*)d_in[7];

  const size_t qkvBytes = (size_t)NB * SEQ * DIM * sizeof(__bf16);   // 32 MiB each
  char* ws = (char*)d_ws;
  __bf16* Qb = (__bf16*)(ws);
  __bf16* Kb = (__bf16*)(ws + qkvBytes);
  __bf16* Vb = (__bf16*)(ws + 2 * qkvBytes);
  float*  S  = (float*) (ws + 3 * qkvBytes);                         // 128 MiB
  __bf16* P  = (__bf16*)(ws + 3 * qkvBytes +
                         (size_t)NB * SEQ * SEQ * sizeof(float));    // 64 MiB

  const dim3 blk(256);
  const float scaleQ = 0.03125f;   // 1/sqrt(1024)

  const dim3 g1(DIM / 128, (NB * SEQ) / 128);                        // (8,128)
  qkv_gemm_kernel<<<g1, blk, 0, stream>>>(x, Wq, bq, Qb, DIM, DIM, scaleQ);
  qkv_gemm_kernel<<<g1, blk, 0, stream>>>(x, Wk, bk, Kb, DIM, DIM, 1.0f);
  qkv_gemm_kernel<<<g1, blk, 0, stream>>>(x, Wv, bv, Vb, DIM, DIM, 1.0f);

  const dim3 g2(SEQ / 128, SEQ / 128, NB);                           // (16,16,8)
  scores_kernel<<<g2, blk, 0, stream>>>(Qb, Kb, S);

  softmax_kernel<<<dim3(NB * SEQ), blk, 0, stream>>>(S, mask, P);

  const dim3 g4(DIM / 128, SEQ / 128, NB);                           // (8,16,8)
  context_kernel<<<g4, blk, 0, stream>>>(P, Vb, (float*)d_out);
}
